// WLN_69123203661939
// MI455X (gfx1250) — matrix operand order repaired and checked by reading, not compiled, not run on hardware
//
#include <hip/hip_runtime.h>

// Problem constants (fixed by the reference harness)
#define V_NODES 100000
#define E_EDGES 250000
#define NODE_IN 82
#define EDGE_IN 12
#define D_OUT   300
#define KP_N    96    // NODE_IN padded to multiple of 32 (WMMA K)
#define KP_E    32    // EDGE_IN padded to multiple of 32
#define DP_K    320   // D padded to multiple of 32 when used as K dim
#define NPAD    304   // D padded to multiple of 16 when used as N dim
#define STR     304   // row stride (floats) for hv / hself / hnbr workspace

typedef __attribute__((ext_vector_type(16))) __bf16 v16bf;
typedef __attribute__((ext_vector_type(8)))  __bf16 v8bf;
typedef __attribute__((ext_vector_type(8)))  float  v8f;

union Frag16 { v16bf v; v8bf h[2]; };

// ---- CDNA5 async global->LDS copy (ASYNCcnt-tracked), 16 bytes per lane ----
__device__ __forceinline__ void async_copy_b128(void* lds_dst, const void* gsrc) {
    // LDS aperture: LDS offset is the low 32 bits of the generic pointer
    unsigned lds = (unsigned)(uintptr_t)lds_dst;
    asm volatile("global_load_async_to_lds_b128 %0, %1, off"
                 :: "v"(lds), "v"(gsrc)
                 : "memory");
}
__device__ __forceinline__ void wait_asynccnt0() {
    asm volatile("s_wait_asynccnt 0x0" ::: "memory");
}

// A fragment (16x32 bf16): lane m=lane&15, half=lane>>4 holds
// K = [kb+half*8 .. +7] and [kb+16+half*8 .. +7]
__device__ __forceinline__ v16bf load_a_rm(const __bf16* base, int stride, int kb, int lane) {
    const int m = lane & 15, half = lane >> 4;
    const __bf16* p = base + m * stride + kb + half * 8;
    Frag16 f;
    f.h[0] = *(const v8bf*)(p);
    f.h[1] = *(const v8bf*)(p + 16);
    return f.v;
}

// B fragment (32x16 bf16): lane n=lane&15 holds K = [kb+half*8 .. +7] and [kb+16+half*8 .. +7] of row (ncol+n)
__device__ __forceinline__ v16bf load_b_rm(const __bf16* W, int stride, int ncol, int kb, int lane) {
    const int n = lane & 15, half = lane >> 4;
    const __bf16* p = W + (size_t)(ncol + n) * stride + kb + half * 8;
    Frag16 f;
    f.h[0] = *(const v8bf*)(p);
    f.h[1] = *(const v8bf*)(p + 16);
    return f.v;
}

// Convert f32 weight (rows x cols, row-major) to bf16 padded (rpad x cpad), zero fill.
__global__ void wln_cvt_pad(const float* __restrict__ src, __bf16* __restrict__ dst,
                            int rows, int cols, int cpad, int total) {
    int i = blockIdx.x * blockDim.x + threadIdx.x;
    if (i >= total) return;
    int r = i / cpad, c = i - r * cpad;
    float v = (r < rows && c < cols) ? src[r * cols + c] : 0.f;
    dst[i] = (__bf16)v;
}

// Fused node pipeline: h = relu(nf @ Wninᵀ) kept in LDS (bf16), then
// hv = h @ Wmsgᵀ and hself = h @ Wselfᵀ in one pass. 32 nodes per block, 8 waves.
__global__ __launch_bounds__(256) void wln_node_fused(
    const float* __restrict__ nfeat,
    const __bf16* __restrict__ Wnin, const __bf16* __restrict__ Wmsg,
    const __bf16* __restrict__ Wself,
    float* __restrict__ hv, float* __restrict__ hself) {
    __shared__ __align__(16) float  s_nf32[32 * NODE_IN];   // raw f32 stage (async)
    __shared__ __align__(16) __bf16 s_nf[32 * KP_N];        // bf16 A tile, K-padded
    __shared__ __align__(16) __bf16 s_h[32 * DP_K];         // bf16 h tile, K-padded
    const int tid  = threadIdx.x;
    const int lane = tid & 31;
    const int wave = tid >> 5;
    const int node0 = blockIdx.x * 32;           // V = 32 * 3125 exactly

    // Phase 0: async-stage the 32x82 f32 feature tile (10496 B = 656 x b128,
    // per-block base is 16B-aligned) and overlap with zeroing the h tile.
    {
        const char* gbase = (const char*)(nfeat + (size_t)node0 * NODE_IN);
        const int nchunks = (32 * NODE_IN * 4) / 16;      // 656
        for (int i = tid; i < nchunks; i += 256)
            async_copy_b128((char*)s_nf32 + i * 16, gbase + i * 16);
    }
    for (int i = tid; i < 32 * DP_K; i += 256) s_h[i] = (__bf16)0.f;
    wait_asynccnt0();
    __syncthreads();

    // Phase 1: convert staged f32 -> zero-padded bf16 A tile
    for (int i = tid; i < 32 * KP_N; i += 256) {
        int r = i / KP_N, c = i - r * KP_N;
        float v = (c < NODE_IN) ? s_nf32[r * NODE_IN + c] : 0.f;
        s_nf[i] = (__bf16)v;
    }
    __syncthreads();

    const int mt = wave >> 2;                    // M tile: 0..1 (16 nodes each)
    const int ns = wave & 3;                     // N slot: 0..3
    const int n  = lane & 15, half = lane >> 4;

    // GEMM1: 16x16 output tiles over N (19 tiles cover 304), K = 96 (3 WMMA steps)
    const __bf16* a1 = s_nf + mt * 16 * KP_N;
    for (int nt = ns; nt < 19; nt += 4) {
        v8f acc = {};
        #pragma unroll
        for (int ks = 0; ks < KP_N / 32; ++ks) {
            v16bf a = load_a_rm(a1, KP_N, ks * 32, lane);
            v16bf b = load_b_rm(Wnin, KP_N, nt * 16, ks * 32, lane);
            acc = __builtin_amdgcn_wmma_f32_16x16x32_bf16(false, a, false, b,
                                                          (short)0, acc, false, false);
        }
        const int col = nt * 16 + n;
        #pragma unroll
        for (int r = 0; r < 8; ++r) {            // C layout: row = r + 8*half
            float v = acc[r];
            s_h[(mt * 16 + r + 8 * half) * DP_K + col] = (__bf16)(v > 0.f ? v : 0.f);
        }
    }
    __syncthreads();

    // GEMM2 (dual output): K = 320 (10 WMMA steps each for hv and hself)
    const __bf16* a2 = s_h + mt * 16 * DP_K;
    for (int nt = ns; nt < 19; nt += 4) {
        v8f accV = {}, accS = {};
        #pragma unroll
        for (int ks = 0; ks < DP_K / 32; ++ks) {
            v16bf a  = load_a_rm(a2, DP_K, ks * 32, lane);
            v16bf bv = load_b_rm(Wmsg,  DP_K, nt * 16, ks * 32, lane);
            accV = __builtin_amdgcn_wmma_f32_16x16x32_bf16(false, a, false, bv,
                                                           (short)0, accV, false, false);
            v16bf bs = load_b_rm(Wself, DP_K, nt * 16, ks * 32, lane);
            accS = __builtin_amdgcn_wmma_f32_16x16x32_bf16(false, a, false, bs,
                                                           (short)0, accS, false, false);
        }
        const int col = nt * 16 + n;
        #pragma unroll
        for (int r = 0; r < 8; ++r) {
            size_t row = (size_t)(node0 + mt * 16 + r + 8 * half);
            hv[row * STR + col]    = accV[r];
            hself[row * STR + col] = accS[r];
        }
    }
}

// Fused edge pipeline: he tile via one K=32 bf16 WMMA per N-tile, then
// m = he * hv[src] (coalesced L2 gather) and atomic scatter-add into hnbr[dst].
__global__ __launch_bounds__(128) void wln_edge_fused(
    const float* __restrict__ efeat, const int* __restrict__ src,
    const int* __restrict__ dst, const __bf16* __restrict__ Wedge,
    const float* __restrict__ hv, float* __restrict__ hnbr) {
    __shared__ __align__(16) float  s_ef32[16 * EDGE_IN];   // raw f32 stage (async)
    __shared__ __align__(16) __bf16 s_ef[16 * KP_E];
    __shared__ int s_src[16], s_dst[16];
    const int tid  = threadIdx.x;
    const int lane = tid & 31;
    const int wave = tid >> 5;
    const int e0 = blockIdx.x * 16;              // E = 16 * 15625 exactly

    // Async-stage the 16x12 f32 edge-feature tile (768 B = 48 x b128, base aligned)
    {
        const char* gbase = (const char*)(efeat + (size_t)e0 * EDGE_IN);
        if (tid < (16 * EDGE_IN * 4) / 16)
            async_copy_b128((char*)s_ef32 + tid * 16, gbase + tid * 16);
    }
    if (tid < 16) { s_src[tid] = src[e0 + tid]; s_dst[tid] = dst[e0 + tid]; }
    wait_asynccnt0();
    __syncthreads();

    for (int i = tid; i < 16 * KP_E; i += 128) {
        int r = i / KP_E, c = i - r * KP_E;
        float v = (c < EDGE_IN) ? s_ef32[r * EDGE_IN + c] : 0.f;
        s_ef[i] = (__bf16)v;
    }
    __syncthreads();

    const int n = lane & 15, half = lane >> 4;
    for (int nt = wave; nt < 19; nt += 4) {
        v8f he = {};
        v16bf a = load_a_rm(s_ef, KP_E, 0, lane);
        v16bf b = load_b_rm(Wedge, KP_E, nt * 16, 0, lane);
        he = __builtin_amdgcn_wmma_f32_16x16x32_bf16(false, a, false, b,
                                                     (short)0, he, false, false);
        const int col = nt * 16 + n;
        #pragma unroll
        for (int r = 0; r < 8; ++r) {
            const int el = r + 8 * half;         // edge index within tile
            float m = he[r] * hv[(size_t)s_src[el] * STR + col];
            atomicAdd(&hnbr[(size_t)s_dst[el] * STR + col], m);
        }
    }
}

// out[v, :] = hnbr[v, :] * hself[v, :], fully float4-vectorized:
// 300 floats = 75 float4 per dense row; STR = 304 floats = 76 float4 per ws row.
__global__ void wln_final_mul(const float4* __restrict__ hnbr4,
                              const float4* __restrict__ hself4,
                              float4* __restrict__ out4, int total4) {
    int i = blockIdx.x * blockDim.x + threadIdx.x;
    if (i >= total4) return;
    int v = i / 75;
    int q = i - v * 75;
    float4 a = hnbr4[(size_t)v * 76 + q];
    float4 b = hself4[(size_t)v * 76 + q];
    float4 o;
    o.x = a.x * b.x; o.y = a.y * b.y; o.z = a.z * b.z; o.w = a.w * b.w;
    out4[i] = o;
}

extern "C" void kernel_launch(void* const* d_in, const int* in_sizes, int n_in,
                              void* d_out, int out_size, void* d_ws, size_t ws_size,
                              hipStream_t stream) {
    const float* node_feats = (const float*)d_in[0];
    const float* edge_feats = (const float*)d_in[1];
    const int*   src        = (const int*)d_in[2];
    const int*   dst        = (const int*)d_in[3];
    const float* W_node_in  = (const float*)d_in[4];
    const float* W_node_msg = (const float*)d_in[5];
    const float* W_edge_msg = (const float*)d_in[6];
    const float* W_self     = (const float*)d_in[7];
    float* out = (float*)d_out;

    // Carve workspace (~366 MB): bf16 weights + hv/hself/hnbr (stride 304 floats)
    char*  ws  = (char*)d_ws;
    size_t off = 0;
    auto carve = [&](size_t bytes) {
        char* p = ws + off;
        off += (bytes + 255) & ~(size_t)255;
        return p;
    };
    __bf16* Wnin_b  = (__bf16*)carve((size_t)NPAD * KP_N * sizeof(__bf16));
    __bf16* Wmsg_b  = (__bf16*)carve((size_t)NPAD * DP_K * sizeof(__bf16));
    __bf16* Wself_b = (__bf16*)carve((size_t)NPAD * DP_K * sizeof(__bf16));
    __bf16* Wedge_b = (__bf16*)carve((size_t)NPAD * KP_E * sizeof(__bf16));
    float* hv    = (float*)carve((size_t)V_NODES * STR * sizeof(float));
    float* hself = (float*)carve((size_t)V_NODES * STR * sizeof(float));
    float* hnbr  = (float*)carve((size_t)V_NODES * STR * sizeof(float));
    (void)in_sizes; (void)n_in; (void)out_size; (void)ws_size;

    // Weight conversion to padded bf16 (tiny; L2-resident afterwards)
    {
        int t0 = NPAD * KP_N;
        wln_cvt_pad<<<(t0 + 255) / 256, 256, 0, stream>>>(W_node_in, Wnin_b, D_OUT, NODE_IN, KP_N, t0);
        int t1 = NPAD * DP_K;
        wln_cvt_pad<<<(t1 + 255) / 256, 256, 0, stream>>>(W_node_msg, Wmsg_b, D_OUT, D_OUT, DP_K, t1);
        wln_cvt_pad<<<(t1 + 255) / 256, 256, 0, stream>>>(W_self, Wself_b, D_OUT, D_OUT, DP_K, t1);
        int t3 = NPAD * KP_E;
        wln_cvt_pad<<<(t3 + 255) / 256, 256, 0, stream>>>(W_edge_msg, Wedge_b, D_OUT, EDGE_IN, KP_E, t3);
    }

    // Zero the scatter accumulator (graph-capture-safe async memset)
    hipMemsetAsync(hnbr, 0, (size_t)V_NODES * STR * sizeof(float), stream);

    // Fused node GEMMs: 3125 blocks x 256 threads (32 nodes each)
    wln_node_fused<<<V_NODES / 32, 256, 0, stream>>>(node_feats, Wnin_b, Wmsg_b, Wself_b, hv, hself);

    // Fused edge GEMM + gather-multiply + atomic scatter: 15625 blocks x 128
    wln_edge_fused<<<E_EDGES / 16, 128, 0, stream>>>(edge_feats, src, dst, Wedge_b, hv, hnbr);

    // Final elementwise product into dense (V, 300) output, b128 loads/stores
    int total4 = V_NODES * (D_OUT / 4);
    wln_final_mul<<<(total4 + 255) / 256, 256, 0, stream>>>(
        (const float4*)hnbr, (const float4*)hself, (float4*)out, total4);
}
